// GAT_33500744909485
// MI455X (gfx1250) — matrix operand order repaired and checked by reading, not compiled, not run on hardware
//
#include <hip/hip_runtime.h>
#include <hip/hip_bf16.h>

// ---------------------------------------------------------------------------
// GAT (2-layer, 8-head) for gfx1250.
// GEMMs via v_wmma_f32_16x16x32_bf16 with pre-packed bf16 operands (pure
// b128 loads + wmma in the hot loop, 4 N-tiles per wave). Edge phase via
// coalesced gather + global_atomic_add_f32 scatter; per-head global max via
// LDS + global atomicMax on an order-preserving uint encoding.
// ---------------------------------------------------------------------------

typedef __attribute__((ext_vector_type(16))) __bf16 v16bf;
typedef __attribute__((ext_vector_type(8)))  __bf16 v8bf;
typedef __attribute__((ext_vector_type(8)))  float  v8f;

#define GAT_HID     64
#define GAT_EA      16
#define GAT_ASTRIDE 144      // 2*64 + 16
#define GAT_MAXH    8
#define NEG_INF_ORD 0x007FFFFFu   // ford(-inf)

__device__ __forceinline__ __bf16 f2bf(float f) {
  unsigned u = __float_as_uint(f);
  unsigned r = u + 0x7FFFu + ((u >> 16) & 1u);   // round-to-nearest-even
  unsigned short h = (unsigned short)(r >> 16);
  return __builtin_bit_cast(__bf16, h);
}

__device__ __forceinline__ float elu1f(float x)  { return x > 0.f ? x : __expf(x) - 1.f; }
__device__ __forceinline__ float lrelu(float x)  { return x >= 0.f ? x : 0.01f * x; }
__device__ __forceinline__ float dot4(float4 a, float4 b) {
  return a.x * b.x + a.y * b.y + a.z * b.z + a.w * b.w;
}
// order-preserving float <-> uint for atomicMax
__device__ __forceinline__ unsigned ford(float f) {
  unsigned u = __float_as_uint(f);
  return (u & 0x80000000u) ? ~u : (u | 0x80000000u);
}
__device__ __forceinline__ float funord(unsigned u) {
  return __uint_as_float((u & 0x80000000u) ? (u & 0x7FFFFFFFu) : ~u);
}

struct BfPair { v8bf lo; v8bf hi; };
__device__ __forceinline__ v16bf combine16(v8bf lo, v8bf hi) {
  BfPair p; p.lo = lo; p.hi = hi;
  return __builtin_bit_cast(v16bf, p);
}

// --------------------------- utility fills ---------------------------------
__global__ void GAT_fill_f32(float* p, long long n, float v) {
  long long i = (long long)blockIdx.x * blockDim.x + threadIdx.x;
  if (i < n) p[i] = v;
}
__global__ void GAT_fill_u32(unsigned* p, int n, unsigned v) {
  int i = blockIdx.x * blockDim.x + threadIdx.x;
  if (i < n) p[i] = v;
}

// fp32 -> bf16 row-major (A operand prep)
__global__ void GAT_f32_to_bf16(const float* __restrict__ s, __bf16* __restrict__ d,
                                long long n) {
  long long i = (long long)blockIdx.x * blockDim.x + threadIdx.x;
  if (i < n) d[i] = f2bf(s[i]);
}

// Repack W_heads [8][64][64] -> Wcat [64][512] row-major (head h -> cols h*64..)
__global__ void GAT_repack_W(const float* __restrict__ Wh, float* __restrict__ Wcat) {
  int idx = blockIdx.x * blockDim.x + threadIdx.x;
  if (idx >= 64 * 512) return;
  int i = idx >> 9, c = idx & 511;
  int h = c >> 6, j = c & 63;
  Wcat[idx] = Wh[h * 4096 + i * 64 + j];
}

// Pack B [K x N] f32 row-major into per-(ktile, ntile, lane) bf16 fragments:
// fragment element e of lane l for tile (kt, tn) = B[kt*32 + (l>=16?16:0) + e][tn*16 + (l&15)]
// -> 16 contiguous bf16 (32B) per lane, contiguous k order (k = 16*(l/16) + e), same order as the A loads.
__global__ void GAT_pack_B(const float* __restrict__ B, __bf16* __restrict__ Bp,
                           int K, int N) {
  int ntiles = N >> 4, ktiles = K >> 5;
  int idx = blockIdx.x * blockDim.x + threadIdx.x;
  if (idx >= ktiles * ntiles * 32) return;
  int lane = idx & 31;
  int tile = idx >> 5;
  int tn = tile % ntiles, kt = tile / ntiles;
  int col  = tn * 16 + (lane & 15);
  int krow = kt * 32 + ((lane >> 4) << 4);   // +0 or +16
  __bf16* dst = Bp + (size_t)idx * 16;
#pragma unroll
  for (int e = 0; e < 16; ++e) dst[e] = f2bf(B[(size_t)(krow + e) * N + col]);
}

// --------------------------- WMMA GEMM -------------------------------------
// C[M x N] = A[M x K] @ B[K x N]; A bf16 row-major, B pre-packed fragments.
// One wave computes a 16 x 64 strip (4 N-tiles) so each A fragment feeds 4
// WMMAs. M % 16 == 0, N % 64 == 0, K % 32 == 0.
template <int K, int N>
__global__ void GAT_wmma_gemm_bf16(const __bf16* __restrict__ A,
                                   const __bf16* __restrict__ Bp,
                                   float* __restrict__ C, int M) {
  constexpr int NTILES = N >> 4;
  constexpr int NTG    = NTILES >> 2;       // groups of 4 n-tiles
  const int total = (M >> 4) * NTG;
  const int wave  = (int)((blockIdx.x * blockDim.x + threadIdx.x) >> 5);
  if (wave >= total) return;                // whole-wave uniform exit
  const int lane = threadIdx.x & 31;
  const int tm = wave / NTG, tg = wave % NTG;
  const int half = lane >> 4;
  const int l16  = lane & 15;

  const __bf16* Ab = A + (size_t)(tm * 16 + l16) * K + half * 16;
  const __bf16* Bb = Bp + ((size_t)tg * 4 * 32 + lane) * 16;

  v8f acc0 = {}, acc1 = {}, acc2 = {}, acc3 = {};
#pragma unroll
  for (int kt = 0; kt < (K >> 5); ++kt) {
    v8bf alo = *(const v8bf*)(Ab + kt * 32);       // K = base+0..7   (b128)
    v8bf ahi = *(const v8bf*)(Ab + kt * 32 + 8);   // K = base+8..15  (b128)
    v16bf af = combine16(alo, ahi);
    const __bf16* Bk = Bb + (size_t)kt * NTILES * 32 * 16;
    v16bf b0 = *(const v16bf*)(Bk);
    v16bf b1 = *(const v16bf*)(Bk + 32 * 16);
    v16bf b2 = *(const v16bf*)(Bk + 2 * 32 * 16);
    v16bf b3 = *(const v16bf*)(Bk + 3 * 32 * 16);
    acc0 = __builtin_amdgcn_wmma_f32_16x16x32_bf16(false, af, false, b0, (short)0, acc0, false, false);
    acc1 = __builtin_amdgcn_wmma_f32_16x16x32_bf16(false, af, false, b1, (short)0, acc1, false, false);
    acc2 = __builtin_amdgcn_wmma_f32_16x16x32_bf16(false, af, false, b2, (short)0, acc2, false, false);
    acc3 = __builtin_amdgcn_wmma_f32_16x16x32_bf16(false, af, false, b3, (short)0, acc3, false, false);
  }
  // C/D layout: VGPR r -> row tm*16 + half*8 + r, col = tile_col + l16
  float* Cr = C + (size_t)(tm * 16 + half * 8) * N + tg * 64 + l16;
#pragma unroll
  for (int r = 0; r < 8; ++r) {
    Cr[(size_t)r * N]      = acc0[r];
    Cr[(size_t)r * N + 16] = acc1[r];
    Cr[(size_t)r * N + 32] = acc2[r];
    Cr[(size_t)r * N + 48] = acc3[r];
  }
}

// ---------------- per-node attention dot products --------------------------
__global__ void GAT_node_dots(const float* __restrict__ H, const float* __restrict__ a,
                              int nh, float* __restrict__ ssrc, float* __restrict__ stgt,
                              int N) {
  int i = blockIdx.x * blockDim.x + threadIdx.x;
  if (i >= N * nh) return;
  int n = i / nh, h = i % nh;
  int ldh = nh * GAT_HID;
  const float4* Hr = (const float4*)(H + (long long)n * ldh + h * GAT_HID);
  const float4* As = (const float4*)(a + h * GAT_ASTRIDE);
  const float4* At = (const float4*)(a + h * GAT_ASTRIDE + GAT_HID);
  float s0 = 0.f, s1 = 0.f;
#pragma unroll
  for (int q = 0; q < GAT_HID / 4; ++q) {
    float4 hv = Hr[q];
    s0 += dot4(hv, As[q]);
    s1 += dot4(hv, At[q]);
  }
  ssrc[i] = s0;
  stgt[i] = s1;
}

// ---------------- per-edge scores + per-head global max --------------------
__global__ void GAT_edge_scores(const float* __restrict__ ssrc, const float* __restrict__ stgt,
                                const float* __restrict__ ea, const float* __restrict__ a,
                                int nh, const int* __restrict__ src, const int* __restrict__ tgt,
                                float* __restrict__ score, unsigned* __restrict__ gmax, int E) {
  __shared__ unsigned smax[GAT_MAXH];
  int tid = threadIdx.x;
  if (tid < nh) smax[tid] = NEG_INF_ORD;
  __syncthreads();
  int e = blockIdx.x * blockDim.x + tid;
  if (e < E) {
    int s = src[e], t = tgt[e];
    const float4* ev = (const float4*)(ea + (long long)e * GAT_EA);
    float4 e0 = ev[0], e1 = ev[1], e2 = ev[2], e3 = ev[3];
    for (int h = 0; h < nh; ++h) {
      const float4* a4 = (const float4*)(a + h * GAT_ASTRIDE + 2 * GAT_HID);
      float d = dot4(e0, a4[0]) + dot4(e1, a4[1]) + dot4(e2, a4[2]) + dot4(e3, a4[3]);
      float sc = lrelu(ssrc[(long long)s * nh + h] + stgt[(long long)t * nh + h] + d);
      score[(long long)e * nh + h] = sc;
      atomicMax(&smax[h], ford(sc));            // LDS atomic
    }
  }
  __syncthreads();
  if (tid < nh) atomicMax(&gmax[tid], smax[tid]);
}

// ---------------- exp(score - max) and segment-sum denominator -------------
__global__ void GAT_edge_expdenom(float* __restrict__ score, const unsigned* __restrict__ gmax,
                                  float* __restrict__ denom, const int* __restrict__ src,
                                  int nh, int E) {
  int e = blockIdx.x * blockDim.x + threadIdx.x;
  if (e >= E) return;
  int s = src[e];
  for (int h = 0; h < nh; ++h) {
    float m  = funord(gmax[h]);
    float ex = __expf(score[(long long)e * nh + h] - m);
    score[(long long)e * nh + h] = ex;
    atomicAdd(&denom[(long long)s * nh + h], ex);
  }
}

// ---------------- alpha-weighted scatter aggregation -----------------------
// block = (64, 4): x = feature dim (coalesced), y = edge-in-block; heads looped
__global__ void GAT_aggregate(const float* __restrict__ H, const float* __restrict__ exps,
                              const float* __restrict__ denom, const int* __restrict__ src,
                              const int* __restrict__ tgt, float* __restrict__ Hp,
                              int nh, int E) {
  int d = threadIdx.x;
  int e = blockIdx.x * blockDim.y + threadIdx.y;
  if (e >= E) return;
  int s = src[e], t = tgt[e];
  int ldh = nh * GAT_HID;
  for (int h = 0; h < nh; ++h) {
    float alpha = exps[(long long)e * nh + h] / (denom[(long long)s * nh + h] + 1e-16f);
    float v = H[(long long)t * ldh + h * GAT_HID + d];
    atomicAdd(&Hp[(long long)s * ldh + h * GAT_HID + d], alpha * v);
  }
}

// elu(elu(x)) between layers, fused with bf16 conversion for the next GEMM
__global__ void GAT_elu2_to_bf16(const float* __restrict__ p, __bf16* __restrict__ q,
                                 long long n) {
  long long i = (long long)blockIdx.x * blockDim.x + threadIdx.x;
  if (i < n) q[i] = f2bf(elu1f(elu1f(p[i])));
}

// fused elu + row-wise log_softmax over 64 cols; one wave32 per node
__global__ void GAT_elu_logsoftmax(const float* __restrict__ Hp, float* __restrict__ out, int N) {
  int node = (int)((blockIdx.x * blockDim.x + threadIdx.x) >> 5);
  int lane = threadIdx.x & 31;
  if (node >= N) return;
  const float* r = Hp + (long long)node * 64;
  float v0 = elu1f(r[lane]), v1 = elu1f(r[lane + 32]);
  float m = fmaxf(v0, v1);
  for (int o = 16; o > 0; o >>= 1) m = fmaxf(m, __shfl_xor(m, o, 32));
  float ssum = __expf(v0 - m) + __expf(v1 - m);
  for (int o = 16; o > 0; o >>= 1) ssum += __shfl_xor(ssum, o, 32);
  float lse = m + __logf(ssum);
  out[(long long)node * 64 + lane]      = v0 - lse;
  out[(long long)node * 64 + lane + 32] = v1 - lse;
}

// ---------------------------------------------------------------------------
extern "C" void kernel_launch(void* const* d_in, const int* in_sizes, int n_in,
                              void* d_out, int out_size, void* d_ws, size_t ws_size,
                              hipStream_t stream) {
  const float* X         = (const float*)d_in[0];
  const float* edge_attr = (const float*)d_in[1];
  const float* W_heads   = (const float*)d_in[2];
  const float* a_heads   = (const float*)d_in[3];
  const float* W_out     = (const float*)d_in[4];
  const float* a_out     = (const float*)d_in[5];
  const int*   ei        = (const int*)d_in[6];

  const int N = in_sizes[0] / 64;
  const int E = in_sizes[6] / 2;
  const int* src = ei;
  const int* tgt = ei + E;

  // workspace (floats, 256B-aligned bump allocator)
  float* ws = (float*)d_ws;
  size_t off = 0;
  auto take = [&](size_t nf) { float* p = ws + off; off += (nf + 63) & ~(size_t)63; return p; };
  float* Wcat = take(64 * 512);
  float* H1   = take((size_t)N * 512);
  float* ss1  = take((size_t)N * 8);
  float* st1  = take((size_t)N * 8);
  float* sc1  = take((size_t)E * 8);
  float* den1 = take((size_t)N * 8);
  float* Hp1  = take((size_t)N * 512);
  float* H2   = take((size_t)N * 64);
  float* ss2  = take((size_t)N);
  float* st2  = take((size_t)N);
  float* sc2  = take((size_t)E);
  float* den2 = take((size_t)N);
  float* Hp2  = take((size_t)N * 64);
  unsigned* gmax1 = (unsigned*)take(64);
  unsigned* gmax2 = (unsigned*)take(64);
  __bf16* A1bf = (__bf16*)take((size_t)N * 32);    // N*64 bf16
  __bf16* A2bf = (__bf16*)take((size_t)N * 256);   // N*512 bf16
  __bf16* B1p  = (__bf16*)take(64 * 512 / 2);      // 64*512 bf16
  __bf16* B2p  = (__bf16*)take(512 * 64 / 2);      // 512*64 bf16
  if (off * sizeof(float) > ws_size) return;

  // init accumulators (every call; harness does not re-zero between replays)
  long long nHp1 = (long long)N * 512, nHp2 = (long long)N * 64;
  GAT_fill_f32<<<(int)((N * 8 + 255) / 256), 256, 0, stream>>>(den1, N * 8, 0.f);
  GAT_fill_f32<<<(int)((nHp1 + 255) / 256), 256, 0, stream>>>(Hp1, nHp1, 0.f);
  GAT_fill_f32<<<(int)((N + 255) / 256), 256, 0, stream>>>(den2, N, 0.f);
  GAT_fill_f32<<<(int)((nHp2 + 255) / 256), 256, 0, stream>>>(Hp2, nHp2, 0.f);
  GAT_fill_u32<<<1, 32, 0, stream>>>(gmax1, 8, NEG_INF_ORD);
  GAT_fill_u32<<<1, 32, 0, stream>>>(gmax2, 1, NEG_INF_ORD);

  // operand prep: bf16 A copies + packed-fragment B copies
  GAT_repack_W<<<(64 * 512 + 255) / 256, 256, 0, stream>>>(W_heads, Wcat);
  GAT_f32_to_bf16<<<(int)(((long long)N * 64 + 255) / 256), 256, 0, stream>>>(X, A1bf, (long long)N * 64);
  GAT_pack_B<<<((64 / 32) * (512 / 16) * 32 + 255) / 256, 256, 0, stream>>>(Wcat, B1p, 64, 512);
  GAT_pack_B<<<((512 / 32) * (64 / 16) * 32 + 255) / 256, 256, 0, stream>>>(W_out, B2p, 512, 64);

  dim3 aggB(64, 4);

  // ---- layer 1 (8 heads fused): H1 = X @ Wcat ----
  int waves1 = (N / 16) * (512 / 64);
  GAT_wmma_gemm_bf16<64, 512><<<(waves1 + 7) / 8, 256, 0, stream>>>(A1bf, B1p, H1, N);
  GAT_node_dots<<<(N * 8 + 255) / 256, 256, 0, stream>>>(H1, a_heads, 8, ss1, st1, N);
  GAT_edge_scores<<<(E + 255) / 256, 256, 0, stream>>>(ss1, st1, edge_attr, a_heads, 8,
                                                       src, tgt, sc1, gmax1, E);
  GAT_edge_expdenom<<<(E + 255) / 256, 256, 0, stream>>>(sc1, gmax1, den1, src, 8, E);
  GAT_aggregate<<<(E + 3) / 4, aggB, 0, stream>>>(H1, sc1, den1, src, tgt, Hp1, 8, E);
  GAT_elu2_to_bf16<<<(int)((nHp1 + 255) / 256), 256, 0, stream>>>(Hp1, A2bf, nHp1);

  // ---- layer 2 (single head): H2 = x @ W_out ----
  int waves2 = (N / 16) * (64 / 64);
  GAT_wmma_gemm_bf16<512, 64><<<(waves2 + 7) / 8, 256, 0, stream>>>(A2bf, B2p, H2, N);
  GAT_node_dots<<<(N + 255) / 256, 256, 0, stream>>>(H2, a_out, 1, ss2, st2, N);
  GAT_edge_scores<<<(E + 255) / 256, 256, 0, stream>>>(ss2, st2, edge_attr, a_out, 1,
                                                       src, tgt, sc2, gmax2, E);
  GAT_edge_expdenom<<<(E + 255) / 256, 256, 0, stream>>>(sc2, gmax2, den2, src, 1, E);
  GAT_aggregate<<<(E + 3) / 4, aggB, 0, stream>>>(H2, sc2, den2, src, tgt, Hp2, 1, E);

  // ---- elu + log_softmax epilogue ----
  GAT_elu_logsoftmax<<<(N + 7) / 8, 256, 0, stream>>>(Hp2, (float*)d_out, N);
}
